// DeepFoodModel_44006234915659
// MI455X (gfx1250) — compile-verified, matrix operand order checked by reading, not run on hardware
//
#include <hip/hip_runtime.h>
#include <hip/hip_bf16.h>

typedef __attribute__((ext_vector_type(16))) _Float16 v16h;
typedef __attribute__((ext_vector_type(8)))  _Float16 v8h;
typedef __attribute__((ext_vector_type(8)))  float    v8f;
typedef int async_v4i __attribute__((vector_size(16)));   // matches builtin param type

#define IMG 576
#define NUM_CLASSES 101
#define POOLSZ 7
#define MAXP 300
#define FEAT 18
#define FLATD (POOLSZ * POOLSZ * 512)

#define BM 128
#define BN 64
#define BK 32

#if defined(__gfx1250__) && __has_builtin(__builtin_amdgcn_global_load_async_to_lds_b128) && \
    __has_builtin(__builtin_amdgcn_s_wait_asynccnt)
#define HAVE_ASYNC_LDS 1
#else
#define HAVE_ASYNC_LDS 0
#endif

// copy 16B global -> 16B LDS (per-lane addresses); async on gfx1250 when available
__device__ __forceinline__ void copy16_g2l(const _Float16* g, _Float16* l) {
#if HAVE_ASYNC_LDS
    __builtin_amdgcn_global_load_async_to_lds_b128(
        (async_v4i*)g,
        (__attribute__((address_space(3))) async_v4i*)l, 0, 0);
#else
    *(v8h*)l = *(const v8h*)g;
#endif
}

__device__ __forceinline__ void wait_async_lds() {
#if HAVE_ASYNC_LDS
    __builtin_amdgcn_s_wait_asynccnt(0);
#endif
}

// ---------------------------------------------------------------------------
// WMMA 16x16x32 f16 fragment layout (ISA 7.12.2, wave32).
// Lane l, element e: kh=l>>4, v=e>>1, pair=e&1,
//   kk = ((v&4)<<2)+(kh<<3)+((v&3)<<1)+pair
// An aligned 8-wide k-group (kk=8q..8q+7) maps to contiguous e=8*(q>>1)+j in
// lane (x&15)+16*(q&1)  ->  16B async copy + 2x ds_load_b128 per fragment.
// ---------------------------------------------------------------------------

__global__ void cvt_f32_f16(const float* __restrict__ in, _Float16* __restrict__ out, int n) {
    int i = blockIdx.x * blockDim.x + threadIdx.x;
    if (i < n) out[i] = (_Float16)in[i];
}

// f32 weights [K][N] -> f16 swizzled B-fragment layout [KB][NT][32 lanes][16]
__global__ void cvt_swizzle_b(const float* __restrict__ w, _Float16* __restrict__ out,
                              int K, int N, int NT, int total) {
    int i = blockIdx.x * blockDim.x + threadIdx.x;
    if (i >= total) return;
    int e    = i & 15;
    int lane = (i >> 4) & 31;
    int nt   = (i >> 9) % NT;
    int kb   = i / (512 * NT);
    int kh = lane >> 4, v = e >> 1, pair = e & 1;
    int kk = ((v & 4) << 2) + (kh << 3) + ((v & 3) << 1) + pair;
    int k = kb * 32 + kk, n = nt * 16 + (lane & 15);
    out[i] = (k < K && n < N) ? (_Float16)w[k * N + n] : (_Float16)0.0f;
}

// ---------------------------------------------------------------------------
// Implicit-GEMM 3x3 SAME conv. 256 thr = 8 waves; block tile 128x64; each wave
// computes a 16x64 strip (4 WMMA accumulators). A double-buffered in LDS via
// async global->LDS copies, B fragments loaded straight from pre-swizzled
// global (L2-resident). cinShift = log2(Cin) (pow2, >=8) else -1 (scalar path).
// ---------------------------------------------------------------------------
__global__ __launch_bounds__(256) void conv3x3_wmma(
    const _Float16* __restrict__ in, const _Float16* __restrict__ Bsw,
    const float* __restrict__ bias, _Float16* __restrict__ out,
    int H, int W, int Cin, int Cout, int NT, int cinShift, int relu)
{
    __shared__ _Float16 Als[2][8][32][16];      // 16 KB ping-pong
    const int M = H * W, N = Cout, Ktot = 9 * Cin;
    const int m0 = blockIdx.x * BM, n0 = blockIdx.y * BN;
    const int tid = threadIdx.x, lane = tid & 31, wv = tid >> 5;
    const int gnt0 = n0 >> 4;
    v8f c0 = {}, c1 = {}, c2 = {}, c3 = {};
    const int nkb = (Ktot + BK - 1) / BK;

    // staging: 2 units per thread (am0 and am0+64), same k-group q0
    const int q0 = tid & 3, am0 = tid >> 2;
    const int s_lane = ((q0 & 1) << 4);
    const int s_e = (q0 >> 1) << 3;
    const int mA0 = m0 + am0, mA1 = mA0 + 64;
    const int yA0 = mA0 / W, xA0 = mA0 - yA0 * W;   // loop-invariant
    const int yA1 = mA1 / W, xA1 = mA1 - yA1 * W;

    auto stage = [&](int kb, int buf) {
        if (cinShift >= 0) {
            int k = kb * BK + q0 * 8;           // Ktot % 32 == 0 here
            int kq = k >> cinShift;             // kernel tap 0..8
            int ci = k & (Cin - 1);
            int ky = kq / 3, kx = kq - ky * 3;
            _Float16* l0 = &Als[buf][am0 >> 4][(am0 & 15) + s_lane][s_e];
            _Float16* l1 = &Als[buf][(am0 >> 4) + 4][(am0 & 15) + s_lane][s_e];
            {
                int iy = yA0 + ky - 1, ix = xA0 + kx - 1;
                if (mA0 < M && iy >= 0 && iy < H && ix >= 0 && ix < W)
                    copy16_g2l(&in[(iy * W + ix) * Cin + ci], l0);
                else
                    *(v8h*)l0 = (v8h){};
            }
            {
                int iy = yA1 + ky - 1, ix = xA1 + kx - 1;
                if (mA1 < M && iy >= 0 && iy < H && ix >= 0 && ix < W)
                    copy16_g2l(&in[(iy * W + ix) * Cin + ci], l1);
                else
                    *(v8h*)l1 = (v8h){};
            }
        } else {
            // first layer (Cin=3): scalar im2col into fragment layout
            for (int j = 0; j < 16; ++j) {
                int idx = tid * 16 + j;
                int am2 = idx >> 5, kk = idx & 31;
                int m = m0 + am2, k = kb * BK + kk;
                _Float16 val = (_Float16)0.0f;
                if (m < M && k < Ktot) {
                    int y = m / W, x = m - y * W;
                    int kq = k / Cin, ci = k - kq * Cin;
                    int ky = kq / 3, kx = kq - ky * 3;
                    int iy = y + ky - 1, ix = x + kx - 1;
                    if (iy >= 0 && iy < H && ix >= 0 && ix < W)
                        val = in[(iy * W + ix) * Cin + ci];
                }
                int e = (((kk >> 4) & 1) << 3) + (((kk >> 1) & 3) << 1) + (kk & 1);
                Als[buf][am2 >> 4][(am2 & 15) + (((kk >> 3) & 1) << 4)][e] = val;
            }
        }
    };

    stage(0, 0);
    wait_async_lds();
    __syncthreads();
    for (int kb = 0; kb < nkb; ++kb) {
        int buf = kb & 1;
        if (kb + 1 < nkb) stage(kb + 1, buf ^ 1);
        const _Float16* bp = Bsw + ((size_t)((size_t)kb * NT + gnt0) * 32 + lane) * 16;
        v16h a  = *(const v16h*)&Als[buf][wv][lane][0];
        v16h b0 = *(const v16h*)bp;
        v16h b1 = *(const v16h*)(bp + 512);
        v16h b2 = *(const v16h*)(bp + 1024);
        v16h b3 = *(const v16h*)(bp + 1536);
        if (kb + 1 < nkb) __builtin_prefetch(bp + (size_t)NT * 512, 0, 1);
        c0 = __builtin_amdgcn_wmma_f32_16x16x32_f16(false, a, false, b0, (short)0, c0, false, false);
        c1 = __builtin_amdgcn_wmma_f32_16x16x32_f16(false, a, false, b1, (short)0, c1, false, false);
        c2 = __builtin_amdgcn_wmma_f32_16x16x32_f16(false, a, false, b2, (short)0, c2, false, false);
        c3 = __builtin_amdgcn_wmma_f32_16x16x32_f16(false, a, false, b3, (short)0, c3, false, false);
        wait_async_lds();
        __syncthreads();
    }

    v8f cc[4] = {c0, c1, c2, c3};
    const int colb = n0 + (lane & 15);
    const int rbase = m0 + wv * 16 + ((lane >> 4) << 3);
#pragma unroll
    for (int t = 0; t < 4; ++t) {
        int col = colb + t * 16;
        if (col < N) {
            float bsv = bias[col];
#pragma unroll
            for (int r = 0; r < 8; ++r) {
                int m = rbase + r;
                if (m < M) {
                    float v = cc[t][r] + bsv;
                    if (relu) v = fmaxf(v, 0.0f);
                    out[m * N + col] = (_Float16)v;
                }
            }
        }
    }
}

// ---------------------------------------------------------------------------
// Generic WMMA GEMM: act(A[MxK] @ B[KxN] + bias). B pre-swizzled. K % 8 == 0.
// Same 128x64 / 4-accumulator / double-buffered / async-staged structure.
// ---------------------------------------------------------------------------
__global__ __launch_bounds__(256) void gemm_wmma(
    const _Float16* __restrict__ A, const _Float16* __restrict__ Bsw,
    const float* __restrict__ bias, _Float16* __restrict__ outh,
    float* __restrict__ outf, int M, int K, int N, int NT, int relu)
{
    __shared__ _Float16 Als[2][8][32][16];
    const int m0 = blockIdx.x * BM, n0 = blockIdx.y * BN;
    const int tid = threadIdx.x, lane = tid & 31, wv = tid >> 5;
    const int gnt0 = n0 >> 4;
    v8f c0 = {}, c1 = {}, c2 = {}, c3 = {};
    const int nkb = (K + BK - 1) / BK;

    const int q0 = tid & 3, am0 = tid >> 2;
    const int s_lane = ((q0 & 1) << 4);
    const int s_e = (q0 >> 1) << 3;
    const int mA0 = m0 + am0, mA1 = mA0 + 64;

    auto stage = [&](int kb, int buf) {
        int k = kb * BK + q0 * 8;
        _Float16* l0 = &Als[buf][am0 >> 4][(am0 & 15) + s_lane][s_e];
        _Float16* l1 = &Als[buf][(am0 >> 4) + 4][(am0 & 15) + s_lane][s_e];
        if (mA0 < M && k < K) copy16_g2l(&A[(size_t)mA0 * K + k], l0);
        else                  *(v8h*)l0 = (v8h){};
        if (mA1 < M && k < K) copy16_g2l(&A[(size_t)mA1 * K + k], l1);
        else                  *(v8h*)l1 = (v8h){};
    };

    stage(0, 0);
    wait_async_lds();
    __syncthreads();
    for (int kb = 0; kb < nkb; ++kb) {
        int buf = kb & 1;
        if (kb + 1 < nkb) stage(kb + 1, buf ^ 1);
        const _Float16* bp = Bsw + ((size_t)((size_t)kb * NT + gnt0) * 32 + lane) * 16;
        v16h a  = *(const v16h*)&Als[buf][wv][lane][0];
        v16h b0 = *(const v16h*)bp;
        v16h b1 = *(const v16h*)(bp + 512);
        v16h b2 = *(const v16h*)(bp + 1024);
        v16h b3 = *(const v16h*)(bp + 1536);
        if (kb + 1 < nkb) __builtin_prefetch(bp + (size_t)NT * 512, 0, 1);
        c0 = __builtin_amdgcn_wmma_f32_16x16x32_f16(false, a, false, b0, (short)0, c0, false, false);
        c1 = __builtin_amdgcn_wmma_f32_16x16x32_f16(false, a, false, b1, (short)0, c1, false, false);
        c2 = __builtin_amdgcn_wmma_f32_16x16x32_f16(false, a, false, b2, (short)0, c2, false, false);
        c3 = __builtin_amdgcn_wmma_f32_16x16x32_f16(false, a, false, b3, (short)0, c3, false, false);
        wait_async_lds();
        __syncthreads();
    }

    v8f cc[4] = {c0, c1, c2, c3};
    const int colb = n0 + (lane & 15);
    const int rbase = m0 + wv * 16 + ((lane >> 4) << 3);
#pragma unroll
    for (int t = 0; t < 4; ++t) {
        int col = colb + t * 16;
        if (col < N) {
            float bsv = bias[col];
#pragma unroll
            for (int r = 0; r < 8; ++r) {
                int m = rbase + r;
                if (m < M) {
                    float v = cc[t][r] + bsv;
                    if (relu) v = fmaxf(v, 0.0f);
                    if (outh) outh[m * N + col] = (_Float16)v;
                    if (outf) outf[m * N + col] = v;
                }
            }
        }
    }
}

// ---------------------------------------------------------------------------
// 2x2 maxpool NHWC f16; optional f32 mirror (feature map)
// ---------------------------------------------------------------------------
__global__ void maxpool2_f16(const _Float16* __restrict__ in, _Float16* __restrict__ out,
                             float* __restrict__ out_f32, int Ho, int Wo, int C) {
    int i = blockIdx.x * blockDim.x + threadIdx.x;
    int tot = Ho * Wo * C;
    if (i >= tot) return;
    int c = i % C, xy = i / C;
    int xo = xy % Wo, yo = xy / Wo;
    int Wi = Wo * 2;
    const _Float16* p = in + ((yo * 2) * Wi + xo * 2) * C + c;
    float a = (float)p[0], b = (float)p[C];
    float d = (float)p[Wi * C], e = (float)p[Wi * C + C];
    float mx = fmaxf(fmaxf(a, b), fmaxf(d, e));
    out[i] = (_Float16)mx;
    if (out_f32) out_f32[i] = mx;
}

// 1x1 conv, small Cout (RPN heads)
__global__ void conv1x1_small(const _Float16* __restrict__ in, const float* __restrict__ w,
                              const float* __restrict__ b, float* __restrict__ out,
                              int M, int K, int N) {
    int i = blockIdx.x * blockDim.x + threadIdx.x;
    if (i >= M * N) return;
    int m = i / N, n = i - m * N;
    float acc = b[n];
    for (int k = 0; k < K; ++k) acc += (float)in[m * K + k] * w[k * N + n];
    out[i] = acc;
}

// per-position softmax + max-prob score (RPN)
__global__ void softmax_rows_max(const float* __restrict__ logits, float* __restrict__ probs,
                                 float* __restrict__ scores, int M, int C) {
    int m = blockIdx.x * blockDim.x + threadIdx.x;
    if (m >= M) return;
    const float* x = logits + m * C;
    float mx = -3.4e38f;
    for (int c = 0; c < C; ++c) mx = fmaxf(mx, x[c]);
    float s = 0.0f;
    for (int c = 0; c < C; ++c) s += __expf(x[c] - mx);
    float inv = 1.0f / s, pmax = 0.0f;
    for (int c = 0; c < C; ++c) {
        float p = __expf(x[c] - mx) * inv;
        probs[m * C + c] = p;
        pmax = fmaxf(pmax, p);
    }
    if (scores) scores[m] = pmax;
}

// top-k via iterative argmax (n <= 512), single block
__global__ __launch_bounds__(512) void topk_kernel(const float* __restrict__ scores, int n, int k,
                                                   int* __restrict__ idx_out) {
    __shared__ float vals[512];
    __shared__ float sv[512];
    __shared__ int   si[512];
    int t = threadIdx.x;
    vals[t] = (t < n) ? scores[t] : -3.4e38f;
    __syncthreads();
    for (int r = 0; r < k; ++r) {
        sv[t] = vals[t]; si[t] = t;
        __syncthreads();
        for (int s = 256; s > 0; s >>= 1) {
            if (t < s) {
                if (sv[t + s] > sv[t] || (sv[t + s] == sv[t] && si[t + s] < si[t])) {
                    sv[t] = sv[t + s]; si[t] = si[t + s];
                }
            }
            __syncthreads();
        }
        if (t == 0) { idx_out[r] = si[0]; vals[si[0]] = -3.4e38f; }
        __syncthreads();
    }
}

__global__ void gather_rois(const float* __restrict__ bbox, const int* __restrict__ idx,
                            float* __restrict__ rois, int k) {
    int i = blockIdx.x * blockDim.x + threadIdx.x;
    if (i < k * 4) rois[i] = bbox[idx[i >> 2] * 4 + (i & 3)];
}

// ROI pool (tf-resize bilinear, half-pixel centers) -> f16 [MAXP][7*7*512]
__global__ __launch_bounds__(512) void roi_pool_kernel(const float* __restrict__ fm,
                                                       const float* __restrict__ rois,
                                                       _Float16* __restrict__ out,
                                                       int Hf, int Wf, int C) {
    __shared__ int   sy0[POOLSZ], sy1[POOLSZ], sx0[POOLSZ], sx1[POOLSZ];
    __shared__ float swy[POOLSZ], swx[POOLSZ];
    int b = blockIdx.x;
    if (threadIdx.x == 0) {
        float r0 = rois[b * 4 + 0], r1 = rois[b * 4 + 1];
        float r2 = rois[b * 4 + 2], r3 = rois[b * 4 + 3];
        int x1 = min(max((int)r0, 0), Wf - 1);
        int y1 = min(max((int)r1, 0), Hf - 1);
        int x2 = min(max((int)r2, x1 + 1), Wf);
        int y2 = min(max((int)r3, y1 + 1), Hf);
        float ny = (float)(y2 - y1), nx = (float)(x2 - x1);
        for (int p = 0; p < POOLSZ; ++p) {
            float sy = ((float)p + 0.5f) * (ny / (float)POOLSZ) - 0.5f;
            sy = fminf(fmaxf(sy, 0.0f), ny - 1.0f) + (float)y1;
            int i0 = (int)floorf(sy);
            sy0[p] = i0; sy1[p] = min(i0 + 1, Hf - 1); swy[p] = sy - (float)i0;
            float sx = ((float)p + 0.5f) * (nx / (float)POOLSZ) - 0.5f;
            sx = fminf(fmaxf(sx, 0.0f), nx - 1.0f) + (float)x1;
            int j0 = (int)floorf(sx);
            sx0[p] = j0; sx1[p] = min(j0 + 1, Wf - 1); swx[p] = sx - (float)j0;
        }
    }
    __syncthreads();
    int tot = POOLSZ * POOLSZ * C;
    for (int i = threadIdx.x; i < tot; i += blockDim.x) {
        int c = i % C, pp = i / C;
        int px = pp % POOLSZ, py = pp / POOLSZ;
        int y0 = sy0[py], y1i = sy1[py], x0 = sx0[px], x1i = sx1[px];
        float wy = swy[py], wx = swx[px];
        float v00 = fm[(y0 * Wf + x0) * C + c];
        float v01 = fm[(y0 * Wf + x1i) * C + c];
        float v10 = fm[(y1i * Wf + x0) * C + c];
        float v11 = fm[(y1i * Wf + x1i) * C + c];
        float v = (1.0f - wy) * (1.0f - wx) * v00 + (1.0f - wy) * wx * v01 +
                  wy * (1.0f - wx) * v10 + wy * wx * v11;
        out[(size_t)b * tot + i] = (_Float16)v;
    }
}

// row softmax (class scores)
__global__ void softmax_rows(const float* __restrict__ logits, float* __restrict__ probs,
                             int M, int C) {
    int m = blockIdx.x * blockDim.x + threadIdx.x;
    if (m >= M) return;
    const float* x = logits + m * C;
    float mx = -3.4e38f;
    for (int c = 0; c < C; ++c) mx = fmaxf(mx, x[c]);
    float s = 0.0f;
    for (int c = 0; c < C; ++c) s += __expf(x[c] - mx);
    float inv = 1.0f / s;
    for (int c = 0; c < C; ++c) probs[m * C + c] = __expf(x[c] - mx) * inv;
}

// ---------------------------------------------------------------------------
// host orchestration
// ---------------------------------------------------------------------------
static const int CFG_H[18] = {64, 64, 0, 128, 128, 0, 256, 256, 256, 0,
                              512, 512, 512, 0, 512, 512, 512, 0};

static inline int ilog2i(int v) { int s = 0; while ((1 << s) < v) ++s; return s; }

extern "C" void kernel_launch(void* const* d_in, const int* in_sizes, int n_in,
                              void* d_out, int out_size, void* d_ws, size_t ws_size,
                              hipStream_t stream) {
    (void)in_sizes; (void)n_in; (void)out_size; (void)ws_size;
    const float* images     = (const float*)d_in[0];
    const float* rpn_conv_w = (const float*)d_in[27];
    const float* rpn_conv_b = (const float*)d_in[28];
    const float* rpn_cls_w  = (const float*)d_in[29];
    const float* rpn_cls_b  = (const float*)d_in[30];
    const float* rpn_bbox_w = (const float*)d_in[31];
    const float* rpn_bbox_b = (const float*)d_in[32];
    const float* fc1_w = (const float*)d_in[33];
    const float* fc1_b = (const float*)d_in[34];
    const float* fc2_w = (const float*)d_in[35];
    const float* fc2_b = (const float*)d_in[36];
    const float* cls_w = (const float*)d_in[37];
    const float* cls_b = (const float*)d_in[38];
    const float* breg_w = (const float*)d_in[39];
    const float* breg_b = (const float*)d_in[40];

    float* out = (float*)d_out;
    float* out_cls   = out;                                  // 300*101
    float* out_breg  = out + MAXP * NUM_CLASSES;             // 300*404
    float* out_rprob = out_breg + MAXP * NUM_CLASSES * 4;    // 324*18
    float* out_rbbox = out_rprob + FEAT * FEAT * 18;         // 324*36

    char* ws = (char*)d_ws;
    size_t off = 0;
    auto bump = [&](size_t bytes) -> void* {
        void* p = ws + off;
        off += (bytes + 255) & ~(size_t)255;
        return p;
    };
    const size_t ACT_MAX = (size_t)IMG * IMG * 64;
    _Float16* actA = (_Float16*)bump(ACT_MAX * 2);
    _Float16* actB = (_Float16*)bump(ACT_MAX * 2);
    _Float16* wbuf = (_Float16*)bump((size_t)FLATD * 4096 * 2);  // fc1 swizzled = max
    float*    fm32 = (float*)bump((size_t)FEAT * FEAT * 512 * 4);
    _Float16* rpn_r = (_Float16*)bump((size_t)FEAT * FEAT * 512 * 2);
    float* rpn_cls_logits = (float*)bump((size_t)FEAT * FEAT * 18 * 4);
    float* scores = (float*)bump(FEAT * FEAT * 4);
    int*   topidx = (int*)bump(MAXP * 4);
    float* rois   = (float*)bump(MAXP * 4 * 4);
    _Float16* h0 = (_Float16*)bump((size_t)MAXP * FLATD * 2);
    _Float16* h1 = (_Float16*)bump((size_t)MAXP * 4096 * 2);
    _Float16* h2 = (_Float16*)bump((size_t)MAXP * 4096 * 2);
    float* cls_logits = (float*)bump((size_t)MAXP * NUM_CLASSES * 4);

    auto swz = [&](const float* src, int K, int N) -> int {
        int NT = ((N + 63) / 64) * 4;
        int KB = (K + 31) / 32;
        int total = KB * NT * 512;
        cvt_swizzle_b<<<dim3((total + 255) / 256), 256, 0, stream>>>(src, wbuf, K, N, NT, total);
        return NT;
    };

    // ---- VGG16 backbone ----
    cvt_f32_f16<<<dim3((IMG * IMG * 3 + 255) / 256), 256, 0, stream>>>(images, actA, IMG * IMG * 3);
    int H = IMG, W = IMG, C = 3, wi = 0, npool = 0;
    _Float16* cur = actA;
    _Float16* nxt = actB;
    for (int li = 0; li < 18; ++li) {
        int c = CFG_H[li];
        if (c == 0) {
            ++npool;
            int Ho = H / 2, Wo = W / 2;
            int tot = Ho * Wo * C;
            float* f32o = (npool == 5) ? fm32 : nullptr;
            maxpool2_f16<<<dim3((tot + 255) / 256), 256, 0, stream>>>(cur, nxt, f32o, Ho, Wo, C);
            H = Ho; W = Wo;
        } else {
            int NT = swz((const float*)d_in[1 + wi], 9 * C, c);
            int cShift = ((C & 7) == 0 && (C & (C - 1)) == 0) ? ilog2i(C) : -1;
            dim3 g((H * W + BM - 1) / BM, c / BN);
            conv3x3_wmma<<<g, 256, 0, stream>>>(cur, wbuf, (const float*)d_in[14 + wi],
                                                nxt, H, W, C, c, NT, cShift, 1);
            C = c; ++wi;
        }
        _Float16* t = cur; cur = nxt; nxt = t;
    }
    // cur = fm (18x18x512 f16); fm32 = f32 copy

    // ---- RPN ----
    {
        int NT = swz(rpn_conv_w, 9 * 512, 512);
        dim3 g((FEAT * FEAT + BM - 1) / BM, 512 / BN);
        conv3x3_wmma<<<g, 256, 0, stream>>>(cur, wbuf, rpn_conv_b, rpn_r,
                                            FEAT, FEAT, 512, 512, NT, ilog2i(512), 1);
    }
    conv1x1_small<<<dim3((FEAT * FEAT * 18 + 255) / 256), 256, 0, stream>>>(
        rpn_r, rpn_cls_w, rpn_cls_b, rpn_cls_logits, FEAT * FEAT, 512, 18);
    conv1x1_small<<<dim3((FEAT * FEAT * 36 + 255) / 256), 256, 0, stream>>>(
        rpn_r, rpn_bbox_w, rpn_bbox_b, out_rbbox, FEAT * FEAT, 512, 36);
    softmax_rows_max<<<dim3((FEAT * FEAT + 127) / 128), 128, 0, stream>>>(
        rpn_cls_logits, out_rprob, scores, FEAT * FEAT, 18);
    topk_kernel<<<1, 512, 0, stream>>>(scores, FEAT * FEAT, MAXP, topidx);
    gather_rois<<<dim3((MAXP * 4 + 255) / 256), 256, 0, stream>>>(out_rbbox, topidx, rois, MAXP);
    roi_pool_kernel<<<MAXP, 512, 0, stream>>>(fm32, rois, h0, FEAT, FEAT, 512);

    // ---- FC head ----
    {
        int NT = swz(fc1_w, FLATD, 4096);
        gemm_wmma<<<dim3((MAXP + BM - 1) / BM, 4096 / BN), 256, 0, stream>>>(
            h0, wbuf, fc1_b, h1, nullptr, MAXP, FLATD, 4096, NT, 1);
    }
    {
        int NT = swz(fc2_w, 4096, 4096);
        gemm_wmma<<<dim3((MAXP + BM - 1) / BM, 4096 / BN), 256, 0, stream>>>(
            h1, wbuf, fc2_b, h2, nullptr, MAXP, 4096, 4096, NT, 1);
    }
    {
        int NT = swz(cls_w, 4096, NUM_CLASSES);
        gemm_wmma<<<dim3((MAXP + BM - 1) / BM, (NUM_CLASSES + BN - 1) / BN), 256, 0, stream>>>(
            h2, wbuf, cls_b, nullptr, cls_logits, MAXP, 4096, NUM_CLASSES, NT, 0);
    }
    softmax_rows<<<dim3((MAXP + 127) / 128), 128, 0, stream>>>(cls_logits, out_cls, MAXP, NUM_CLASSES);
    {
        int NT = swz(breg_w, 4096, NUM_CLASSES * 4);
        gemm_wmma<<<dim3((MAXP + BM - 1) / BM, (NUM_CLASSES * 4 + BN - 1) / BN), 256, 0, stream>>>(
            h2, wbuf, breg_b, nullptr, out_breg, MAXP, 4096, NUM_CLASSES * 4, NT, 0);
    }
}
